// ModifiedTNRDStage_74912819577234
// MI455X (gfx1250) — matrix operand order repaired and checked by reading, not compiled, not run on hardware
//
#include <hip/hip_runtime.h>

typedef __attribute__((ext_vector_type(16))) _Float16 v16h;
typedef __attribute__((ext_vector_type(8)))  float    v8f;

#define H_IMG  256
#define W_IMG  256
#define NFILT  48
#define KTAP   49     // 7x7
#define NBASIS 31

// =====================================================================
// Kernel 1: conv1 as implicit GEMM via v_wmma_f32_16x16x32_f16, fused
// with the 31-basis RBF influence.  One workgroup = one 16x16 spatial
// tile of one batch image; computes all 48 filter channels.
//   O[48 x 256px] = F[48 x 64(pad)] * P[64(pad) x 256px]
// 128 threads = 4 waves; each wave owns 4 pixel rows (4 N-tiles).
// RBF is evaluated with k outer / 24 elements inner so 24 independent
// exp->fma chains are in flight (TRANS co-issues with VALU).
// =====================================================================
__global__ __launch_bounds__(128)
void k1_conv_rbf(const float* __restrict__ u,
                 const float* __restrict__ filters,   // [48*49]
                 const float* __restrict__ rbfw,      // [48*31]
                 float* __restrict__ phi)             // [2*48*256*256]
{
    __shared__ float    u_lds[26 * 24];          // 22x22 halo tile, padded
    __shared__ _Float16 flt_lds[NFILT * 64];     // filters, K padded 49->64
    __shared__ float    w_lds[NFILT * NBASIS];

    const int tid = threadIdx.x;
    const int wg  = blockIdx.x;
    const int b   = wg >> 8;
    const int t   = wg & 255;
    const int ty  = t >> 4, tx = t & 15;
    const int gy0 = ty * 16 - 3, gx0 = tx * 16 - 3;
    const float* ub = u + b * (H_IMG * W_IMG);

    // ---- stage u halo tile (zero-padded SAME borders) ----
    for (int i = tid; i < 26 * 24; i += 128) {
        int r = i / 24, c = i % 24;
        int gy = gy0 + r, gx = gx0 + c;
        float v = 0.0f;
        if (r < 22 && c < 22 && gy >= 0 && gy < H_IMG && gx >= 0 && gx < W_IMG)
            v = ub[gy * W_IMG + gx];
        u_lds[i] = v;
    }
    // ---- stage filters as f16, K padded with zeros ----
    for (int i = tid; i < NFILT * 64; i += 128) {
        int fi = i >> 6, k = i & 63;
        flt_lds[i] = (_Float16)((k < KTAP) ? filters[fi * KTAP + k] : 0.0f);
    }
    // ---- stage RBF weights ----
    for (int i = tid; i < NFILT * NBASIS; i += 128)
        w_lds[i] = rbfw[i];
    __syncthreads();

    const int lane  = tid & 31;
    const int wid   = tid >> 5;
    const int n     = lane & 15;   // M index (A) / pixel column (B,C)
    const int khalf = lane >> 4;

    // ---- A fragments (filters), 16-bit A 16x32, contiguous K per lane half (k = 16h + s) ----
    // lanes 0-15: K = 0..15; lanes 16-31: K = 16..31
    v16h afrag[3][2];
    #pragma unroll
    for (int mt = 0; mt < 3; ++mt) {
        #pragma unroll
        for (int s = 0; s < 2; ++s) {
            #pragma unroll
            for (int j = 0; j < 8; ++j) {
                int K = 32 * s + 16 * khalf + 2 * j;
                afrag[mt][s][2 * j]     = flt_lds[(16 * mt + n) * 64 + K];
                afrag[mt][s][2 * j + 1] = flt_lds[(16 * mt + n) * 64 + K + 1];
            }
        }
    }

    #pragma unroll 1
    for (int rr = 0; rr < 4; ++rr) {
        const int r = wid * 4 + rr;             // pixel row within tile
        // ---- B fragments (im2col patches), 16-bit B 32x16 layout ----
        // lanes 0-15: K = 0..15 ; lanes 16-31: K = 16..31 (VGPR j -> K=2j,2j+1)
        v16h bfrag[2];
        #pragma unroll
        for (int s = 0; s < 2; ++s) {
            #pragma unroll
            for (int j = 0; j < 8; ++j) {
                int K0 = 32 * s + 16 * khalf + 2 * j;
                int K1 = K0 + 1;
                // K in [0,63]: row r + K/7 <= 24 < 26, col n + K%7 <= 21 < 24 (in-bounds)
                float x0 = u_lds[(r + K0 / 7) * 24 + n + K0 % 7];
                float x1 = u_lds[(r + K1 / 7) * 24 + n + K1 % 7];
                bfrag[s][2 * j]     = (_Float16)((K0 < KTAP) ? x0 : 0.0f);
                bfrag[s][2 * j + 1] = (_Float16)((K1 < KTAP) ? x1 : 0.0f);
            }
        }

        // ---- 6 WMMAs: all 48 channels for this pixel row ----
        v8f acc[3];
        #pragma unroll
        for (int mt = 0; mt < 3; ++mt) {
            v8f z = {0.f, 0.f, 0.f, 0.f, 0.f, 0.f, 0.f, 0.f};
            z = __builtin_amdgcn_wmma_f32_16x16x32_f16(
                    false, afrag[mt][0], false, bfrag[0], (short)0, z, false, false);
            acc[mt] = __builtin_amdgcn_wmma_f32_16x16x32_f16(
                    false, afrag[mt][1], false, bfrag[1], (short)0, z, false, false);
        }

        // ---- RBF: k outer (rolled), 24 elements inner (unrolled) ----
        // phi = sum_k w[c,k] * exp2((v-mu_k)^2 * -50*log2(e))
        float sums[3][8];
        #pragma unroll
        for (int mt = 0; mt < 3; ++mt)
            #pragma unroll
            for (int i = 0; i < 8; ++i)
                sums[mt][i] = 0.0f;

        #pragma unroll 1
        for (int k = 0; k < NBASIS; ++k) {
            const float mu = -1.0f + (float)k * (2.0f / 30.0f);
            #pragma unroll
            for (int mt = 0; mt < 3; ++mt) {
                #pragma unroll
                for (int i = 0; i < 8; ++i) {
                    const int c = 16 * mt + 8 * khalf + i;
                    float d = acc[mt][i] - mu;
                    float e = __builtin_amdgcn_exp2f(d * d * (-72.13475204444817f));
                    sums[mt][i] = fmaf(w_lds[c * NBASIS + k], e, sums[mt][i]);
                }
            }
        }

        const int gy = ty * 16 + r, gx = tx * 16 + n;
        #pragma unroll
        for (int mt = 0; mt < 3; ++mt) {
            #pragma unroll
            for (int i = 0; i < 8; ++i) {
                const int c = 16 * mt + 8 * khalf + i;
                phi[((b * NFILT + c) * H_IMG + gy) * W_IMG + gx] = sums[mt][i];
            }
        }
    }
}

// =====================================================================
// Kernel 2: diffusion = sum_c conv2d(phi_c, flip(k_c)), fused with the
// reaction term.  Single output channel (M=1) does not map to WMMA, so
// LDS-tiled VALU conv over 8-channel phi chunks with a 3-halo.
// Interior tiles stage phi via GLOBAL_LOAD_ASYNC_TO_LDS_B32 (ASYNCcnt);
// boundary tiles fall back to bounds-checked VGPR staging.
// 256 threads = one thread per output pixel of a 16x16 tile.
// =====================================================================
__global__ __launch_bounds__(256)
void k2_diffuse_combine(const float* __restrict__ phi,
                        const float* __restrict__ filters,
                        const float* __restrict__ u,
                        const float* __restrict__ f,
                        const float* __restrict__ lam_p,
                        float* __restrict__ out)
{
    __shared__ float p_lds[8 * 22 * 24];   // 8 channels x 22x22 halo tile (row pad 24)
    __shared__ float wf_lds[8 * KTAP];     // spatially flipped filters

    const int tid = threadIdx.x;
    const int wg  = blockIdx.x;
    const int b   = wg >> 8;
    const int t   = wg & 255;
    const int ty  = t >> 4, tx = t & 15;
    const int gy0 = ty * 16 - 3, gx0 = tx * 16 - 3;
    const int py  = tid >> 4, px = tid & 15;
    const int gy  = ty * 16 + py, gx = tx * 16 + px;

    const bool interior = (ty >= 1) && (ty <= 14) && (tx >= 1) && (tx <= 14);
    const float* phib = phi + b * (NFILT * H_IMG * W_IMG);
    // LDS byte address of p_lds (flat LDS aperture: addr[31:0] == LDS offset)
    const unsigned lds_base = (unsigned)(unsigned long long)(&p_lds[0]);

    float acc = 0.0f;
    for (int c0 = 0; c0 < NFILT; c0 += 8) {
        __syncthreads();
        if (interior) {
            // async global->LDS staging: 8 channels x 22x22 dwords
            for (int i = tid; i < 8 * 22 * 22; i += 256) {
                int c   = i / (22 * 22);
                int rem = i % (22 * 22);
                int r = rem / 22, cc = rem % 22;
                unsigned lds_addr = lds_base + (unsigned)((c * 528 + r * 24 + cc) * 4);
                int goff = (((c0 + c) * H_IMG + (gy0 + r)) * W_IMG + (gx0 + cc)) * 4;
                asm volatile("global_load_async_to_lds_b32 %0, %1, %2"
                             :
                             : "v"(lds_addr), "v"(goff), "s"(phib)
                             : "memory");
            }
            asm volatile("s_wait_asynccnt 0" ::: "memory");
        } else {
            for (int i = tid; i < 8 * 22 * 24; i += 256) {
                int c   = i / (22 * 24);
                int rem = i % (22 * 24);
                int r = rem / 24, cc = rem % 24;
                int gyy = gy0 + r, gxx = gx0 + cc;
                float v = 0.0f;
                if (cc < 22 && gyy >= 0 && gyy < H_IMG && gxx >= 0 && gxx < W_IMG)
                    v = phib[((c0 + c) * H_IMG + gyy) * W_IMG + gxx];
                p_lds[i] = v;
            }
        }
        for (int i = tid; i < 8 * KTAP; i += 256) {
            int c = i / KTAP, kk = i % KTAP;
            // flip(dy,dx): index c*49 + (48 - kk)
            wf_lds[i] = filters[(c0 + c) * KTAP + (KTAP - 1 - kk)];
        }
        __syncthreads();
        #pragma unroll 1
        for (int lc = 0; lc < 8; ++lc) {
            const float* pl = &p_lds[lc * 22 * 24];
            const float* wl = &wf_lds[lc * KTAP];
            #pragma unroll
            for (int dy = 0; dy < 7; ++dy)
                #pragma unroll
                for (int dx = 0; dx < 7; ++dx)
                    acc = fmaf(pl[(py + dy) * 24 + px + dx], wl[dy * 7 + dx], acc);
        }
    }

    const int gidx = b * (H_IMG * W_IMG) + gy * W_IMG + gx;
    const float uu  = u[gidx];
    const float ff  = f[gidx];
    const float lam = lam_p[0];
    out[gidx] = uu - acc - lam * (uu - ff);
}

// =====================================================================
extern "C" void kernel_launch(void* const* d_in, const int* in_sizes, int n_in,
                              void* d_out, int out_size, void* d_ws, size_t ws_size,
                              hipStream_t stream) {
    (void)in_sizes; (void)n_in; (void)out_size; (void)ws_size;
    const float* u    = (const float*)d_in[0];   // [2,1,256,256]
    const float* fl   = (const float*)d_in[1];   // [2,1,256,256]
    const float* filt = (const float*)d_in[2];   // [48,1,7,7]
    const float* rbw  = (const float*)d_in[3];   // [48,31]
    const float* lam  = (const float*)d_in[4];   // scalar
    float* out = (float*)d_out;
    float* phi = (float*)d_ws;                   // needs 2*48*256*256*4 = 25.2 MB

    dim3 grid(2 * 16 * 16);                      // batch x 16x16 tiles of 16x16 px
    k1_conv_rbf<<<grid, 128, 0, stream>>>(u, filt, rbw, phi);
    k2_diffuse_combine<<<grid, 256, 0, stream>>>(phi, filt, u, fl, lam, out);
}
